// WCP_31129922961770
// MI455X (gfx1250) — compile-verified, matrix operand order checked by reading, not run on hardware
//
#include <hip/hip_runtime.h>
#include <hip/hip_bf16.h>
#include <cstdint>
#include <cstddef>

// ---------------------------------------------------------------------------
// Types for CDNA5 WMMA (wave32): v_wmma_f32_16x16x32_bf16
// ---------------------------------------------------------------------------
typedef __attribute__((ext_vector_type(16))) __bf16 v16bf;
typedef __attribute__((ext_vector_type(8)))  __bf16 v8bf;
typedef __attribute__((ext_vector_type(8)))  float  v8f;

#define NEGV (-1e10f)

// Problem constants (from reference)
#define C_B    256
#define C_TN   256
#define C_IS   512
#define C_HS   256
#define C_H    128
#define C_NH   32
#define C_TH   16
#define C_NHP  8192

// ---------------------------------------------------------------------------
// Workspace layout (byte offsets; all MB-aligned so 32B vector loads are fine)
// ---------------------------------------------------------------------------
#define OFF_PACK   ((size_t)0)            // packed bf16 weights (~5MB) + biases
#define OFF_BIAS   ((size_t)12 << 20)     // combined f32 biases
#define OFF_STATE  ((size_t)16 << 20)     // hF,cF,hB,cB : 4 x 8192*128 f32
#define OFF_HS     ((size_t)32 << 20)     // wenc_hs [256,32,256] f32 (8MB)
#define OFF_P      ((size_t)40 << 20)     // softmax p [256,32,256] (8MB)
#define OFF_CN     ((size_t)48 << 20)     // c_n (8MB)
#define OFF_Y1     ((size_t)56 << 20)     // c_n @ Wc^T (8MB)
#define OFF_Y2     ((size_t)64 << 20)     // hs @ Whs^T (8MB)
#define OFF_XBF    ((size_t)80 << 20)     // bf16 staging (max 134MB)
#define OFF_GF     ((size_t)224 << 20)    // forward pre-gates (max 268MB)
#define OFF_GB     ((size_t)496 << 20)    // backward pre-gates (max 268MB)
#define OFF_YA     ((size_t)768 << 20)    // layer-0 output (max 134MB)
#define OFF_YB     ((size_t)904 << 20)    // layer-1 output (max 134MB)

// ---------------------------------------------------------------------------
// A-fragment assembly from LDS row-major bf16 tile.
// ISA layout (16-bit A 16x32): lanes 0-15: M=lane, K = kb..kb+7 & 16+kb..16+kb+7
// with kb = 0 for lanes 0-15, kb = 8 for lanes 16-31.
// ---------------------------------------------------------------------------
__device__ __forceinline__ v16bf make_afrag(const __bf16* row_k0, int kb) {
  v8bf lo = *(const v8bf*)(row_k0 + kb);
  v8bf hi = *(const v8bf*)(row_k0 + 16 + kb);
  return __builtin_shufflevector(lo, hi, 0,1,2,3,4,5,6,7,8,9,10,11,12,13,14,15);
}

#define WMMA_BF16(A, Bm, Cm) \
  __builtin_amdgcn_wmma_f32_16x16x32_bf16(false, (A), false, (Bm), (short)0, (Cm), false, false)

// CDNA5 async global->LDS copy (ASYNCcnt-tracked). IOFFSET applies to BOTH the
// global source and LDS destination address, so offset:16 copies chunk #2.
__device__ __forceinline__ void async_copy_32B(unsigned lds_addr, const void* gsrc) {
#if defined(__AMDGCN__)
  asm volatile("global_load_async_to_lds_b128 %0, %1, off"
               :: "v"(lds_addr), "v"(gsrc) : "memory");
  asm volatile("global_load_async_to_lds_b128 %0, %1, off offset:16"
               :: "v"(lds_addr), "v"(gsrc) : "memory");
#else
  (void)lds_addr; (void)gsrc;
#endif
}

__device__ __forceinline__ void wait_async_then_barrier() {
#if defined(__AMDGCN__)
  asm volatile("s_wait_asynccnt 0x0" ::: "memory");
#endif
  __syncthreads();
}

// ---------------------------------------------------------------------------
// Elementwise utility kernels
// ---------------------------------------------------------------------------
__global__ void k_cvt_bf16(const float* __restrict__ in, __bf16* __restrict__ out, size_t n) {
  size_t i = (size_t)blockIdx.x * blockDim.x + threadIdx.x;
  if (i < n) out[i] = (__bf16)in[i];
}

__global__ void k_zero_f32(float* __restrict__ p, size_t n) {
  size_t i = (size_t)blockIdx.x * blockDim.x + threadIdx.x;
  if (i < n) p[i] = 0.0f;
}

__global__ void k_bias2(const float* __restrict__ a, const float* __restrict__ b,
                        float* __restrict__ o, int n) {
  int i = blockIdx.x * blockDim.x + threadIdx.x;
  if (i < n) o[i] = a[i] + b[i];
}

// ---------------------------------------------------------------------------
// Repack W [Nout rows, K cols] (row-major f32, used as B = W^T [K,Nout]) into
// the lane-swizzled bf16 B-fragment layout:
//   packed[((nt*kTiles + kt)*32 + lane)*16 + e]
//   n = nt*16 + (lane&15); kb = (lane>>4)*8
//   k = kt*32 + (e<8 ? kb+e : 16+kb+(e-8))
// so a GEMM wave loads a full 32x16 B tile as one contiguous 32B chunk/lane.
// ---------------------------------------------------------------------------
__global__ void k_pack_w(const float* __restrict__ W, __bf16* __restrict__ out,
                         int K, int Nout) {
  size_t pid = (size_t)blockIdx.x * blockDim.x + threadIdx.x;
  size_t total = (size_t)K * Nout;
  if (pid >= total) return;
  int e      = (int)(pid & 15);
  int lane   = (int)((pid >> 4) & 31);
  int kTiles = K >> 5;
  int kt     = (int)((pid >> 9) % (size_t)kTiles);
  int nt     = (int)(pid / ((size_t)512 * kTiles));
  int n  = nt * 16 + (lane & 15);
  int kb = (lane >> 4) << 3;
  int k  = kt * 32 + (e < 8 ? kb + e : 16 + kb + (e - 8));
  out[pid] = (__bf16)W[(size_t)n * K + k];
}

// ---------------------------------------------------------------------------
// WMMA GEMM: C[M,Ncols] f32 = A[M,K] bf16 row-major  x  Bpacked (+bias[Ncols])
// Block: 128 threads (4 waves). Block tile = 64 rows x 256 cols; each wave
// owns 4 n-tiles (64 cols) x 4 m-tiles = 16 WMMAs per K-step, amortizing the
// async-LDS A staging and the packed-B loads 4x. grid = (M/64, Ncols/256).
// ---------------------------------------------------------------------------
__global__ __launch_bounds__(128) void k_gemm(
    const __bf16* __restrict__ A, const __bf16* __restrict__ Bp,
    const float* __restrict__ bias, float* __restrict__ C,
    int M, int K, int Ncols) {
  __shared__ __align__(16) __bf16 aT[64][32];   // 4KB A tile (64 x 32 bf16)
  const int tid  = threadIdx.x;
  const int wave = tid >> 5, lane = tid & 31;
  const int m0 = blockIdx.x * 64;
  const int c0 = blockIdx.y * 256 + wave * 64;
  const int kTiles = K >> 5;
  const int nt0 = c0 >> 4;
  const int lrow = lane & 15;
  const int kb   = (lane >> 4) << 3;
  const int ldrow = tid >> 1;                  // 64 rows
  const int ldcol = (tid & 1) << 4;            // 2 x 16-elem (32B) chunks/row
  const unsigned ldsDst = (unsigned)(uintptr_t)&aT[ldrow][ldcol];

  v8f acc[4][4] = {};

  for (int kt = 0; kt < kTiles; ++kt) {
    __syncthreads();                           // LDS tile reuse guard
    // Stage A tile via CDNA5 async global->LDS (2 x B128 per thread)
    const __bf16* asrc = A + (size_t)(m0 + ldrow) * K + (kt << 5) + ldcol;
    async_copy_32B(ldsDst, asrc);
    if (kt + 1 < kTiles)
      __builtin_prefetch(asrc + 32, 0, 1);     // global_prefetch_b8 next tile
    wait_async_then_barrier();

    v16bf afr[4];
#pragma unroll
    for (int mt = 0; mt < 4; ++mt)
      afr[mt] = make_afrag(&aT[mt * 16 + lrow][0], kb);

    const __bf16* bb = Bp + ((size_t)nt0 * kTiles + kt) * 512 + (size_t)lane * 16;
    const size_t ntStride = (size_t)kTiles * 512;
#pragma unroll
    for (int nt = 0; nt < 4; ++nt) {
      v16bf b = *(const v16bf*)(bb + (size_t)nt * ntStride);
      acc[0][nt] = WMMA_BF16(afr[0], b, acc[0][nt]);
      acc[1][nt] = WMMA_BF16(afr[1], b, acc[1][nt]);
      acc[2][nt] = WMMA_BF16(afr[2], b, acc[2][nt]);
      acc[3][nt] = WMMA_BF16(afr[3], b, acc[3][nt]);
    }
  }

  // C/D layout: VGPR r, lanes 0-15 -> M=r, lanes 16-31 -> M=8+r; N = lane&15
  const int col  = lane & 15;
  const int mOff = (lane >> 4) << 3;
#pragma unroll
  for (int nt = 0; nt < 4; ++nt) {
    const int c = c0 + nt * 16 + col;
    const float bv = bias ? bias[c] : 0.0f;
#pragma unroll
    for (int mt = 0; mt < 4; ++mt) {
      float* cp = C + (size_t)(m0 + mt * 16 + mOff) * (size_t)Ncols + c;
#pragma unroll
      for (int r = 0; r < 8; ++r)
        cp[(size_t)r * Ncols] = acc[mt][nt][r] + bv;
    }
  }
}

// ---------------------------------------------------------------------------
// Fused LSTM recurrence step: per block 16 rows; grid.y = direction (0 fwd,
// 1 bwd). Computes h_prev @ Whh^T via WMMA (K=128, 512 gate cols spread over
// 8 waves), adds precomputed input gates, applies activations with
// packed-sequence freeze/zero semantics, updates h/c state and writes the
// masked sequence output in-place at the original time index.
// ---------------------------------------------------------------------------
__global__ __launch_bounds__(256) void k_lstm_step(
    const float* __restrict__ gatesF, const float* __restrict__ gatesB,
    const __bf16* __restrict__ WhF, const __bf16* __restrict__ WhB,
    float* __restrict__ hF, float* __restrict__ cF,
    float* __restrict__ hB, float* __restrict__ cB,
    const int* __restrict__ lens, float* __restrict__ y,
    int N, int T, int t) {
  __shared__ __align__(16) __bf16 aT[16][128];
  __shared__ float gt[16][512];
  const int dir = blockIdx.y;
  const float*  gates = dir ? gatesB : gatesF;
  const __bf16* Wh    = dir ? WhB    : WhF;
  float* hS = dir ? hB : hF;
  float* cS = dir ? cB : cF;
  const int r0  = blockIdx.x * 16;
  const int tid = threadIdx.x;

  { // stage h_prev (16x128 f32) -> bf16 LDS, A-tile row-major
    const int row = tid >> 4, cb = (tid & 15) << 3;
    const float* src = hS + (size_t)(r0 + row) * 128 + cb;
#pragma unroll
    for (int j = 0; j < 8; ++j) aT[row][cb + j] = (__bf16)src[j];
  }
  __syncthreads();

  const int wave = tid >> 5, lane = tid & 31;
  const int c0 = wave * 64;                    // 8 waves x 64 cols = 512 gates
  const int lrow = lane & 15, kb = (lane >> 4) << 3;
  v8f acc0 = {}, acc1 = {}, acc2 = {}, acc3 = {};
#pragma unroll
  for (int kt = 0; kt < 4; ++kt) {             // K = 128 = 4 x 32
    v16bf a = make_afrag(&aT[lrow][kt << 5], kb);
    const __bf16* bb = Wh + ((size_t)(c0 >> 4) * 4 + kt) * 512 + (size_t)lane * 16;
    v16bf b0 = *(const v16bf*)(bb);
    v16bf b1 = *(const v16bf*)(bb + 2048);     // nTile stride = kTiles(4)*512
    v16bf b2 = *(const v16bf*)(bb + 4096);
    v16bf b3 = *(const v16bf*)(bb + 6144);
    acc0 = WMMA_BF16(a, b0, acc0);
    acc1 = WMMA_BF16(a, b1, acc1);
    acc2 = WMMA_BF16(a, b2, acc2);
    acc3 = WMMA_BF16(a, b3, acc3);
  }
  const int col = lane & 15, mOff = (lane >> 4) << 3;
#define GSTORE(ACC, NT) do {                                          \
    _Pragma("unroll") for (int r = 0; r < 8; ++r)                     \
      gt[mOff + r][c0 + (NT) * 16 + col] = (ACC)[r];                  \
  } while (0)
  GSTORE(acc0, 0); GSTORE(acc1, 1); GSTORE(acc2, 2); GSTORE(acc3, 3);
#undef GSTORE
  __syncthreads();

  // Activation: thread -> (row, 8 hidden units)
  const int row = tid >> 4;
  const int jb  = (tid & 15) << 3;
  const int n   = r0 + row;
  const int len = lens[n];
  if (t < len) {
    const int idx = dir ? (len - 1 - t) : t;   // bwd indexes original position
    const float* gp = gates + ((size_t)n * T + idx) * 512;
    float* yp = y + ((size_t)n * T + idx) * 256 + dir * 128;
#pragma unroll
    for (int jj = 0; jj < 8; ++jj) {
      const int j = jb + jj;
      const float gi = gp[j]       + gt[row][j];
      const float gf = gp[128 + j] + gt[row][128 + j];
      const float gg = gp[256 + j] + gt[row][256 + j];
      const float go = gp[384 + j] + gt[row][384 + j];
      const float cprev = cS[(size_t)n * 128 + j];
      const float si = 1.0f / (1.0f + __expf(-gi));
      const float sf = 1.0f / (1.0f + __expf(-gf));
      const float so = 1.0f / (1.0f + __expf(-go));
      const float cc = sf * cprev + si * tanhf(gg);
      const float hh = so * tanhf(cc);
      cS[(size_t)n * 128 + j] = cc;            // state frozen when t >= len
      hS[(size_t)n * 128 + j] = hh;
      yp[j] = hh;                              // y pre-zeroed -> masked output
    }
  }
}

// ---------------------------------------------------------------------------
// Header last-state gather: wenc_hs[n,:] = yH[n, l_hpu[n]-1, :]
// (l_hs is constant NH=32, so b_idx = n/32, pos = n%32 -> flat identity)
// ---------------------------------------------------------------------------
__global__ void k_gather_hs(const float* __restrict__ yH,
                            const int* __restrict__ l_hpu,
                            float* __restrict__ hs) {
  size_t gid = (size_t)blockIdx.x * blockDim.x + threadIdx.x;
  if (gid >= (size_t)C_NHP * C_HS) return;
  const int d = (int)(gid & 255);
  const size_t n = gid >> 8;
  const int last = l_hpu[n] - 1;
  hs[gid] = yH[((size_t)n * C_TH + last) * C_HS + d];
}

// ---------------------------------------------------------------------------
// Attention scores + masked softmax. Block = one (b,h) pair, 256 threads
// (one per question token). hs row staged in LDS; block reductions for
// max/sum over tokens with t >= l_n masked to -1e10.
// ---------------------------------------------------------------------------
__global__ __launch_bounds__(256) void k_att_softmax(
    const float* __restrict__ hs, const float* __restrict__ proj,
    const int* __restrict__ l_n, float* __restrict__ p) {
  __shared__ float hrow[256];
  __shared__ float red[256];
  const int b = blockIdx.x >> 5;
  const int h = blockIdx.x & 31;
  const int tid = threadIdx.x;
  hrow[tid] = hs[((size_t)b * 32 + h) * 256 + tid];
  __syncthreads();
  const float* pr = proj + ((size_t)b * 256 + tid) * 256;
  float s = 0.0f;
  for (int d = 0; d < 256; d += 4) {
    const float4 q = *reinterpret_cast<const float4*>(pr + d);
    s += hrow[d] * q.x + hrow[d + 1] * q.y + hrow[d + 2] * q.z + hrow[d + 3] * q.w;
  }
  if (tid >= l_n[b]) s = NEGV;
  red[tid] = s; __syncthreads();
  for (int off = 128; off > 0; off >>= 1) {
    if (tid < off) red[tid] = fmaxf(red[tid], red[tid + off]);
    __syncthreads();
  }
  const float mx = red[0]; __syncthreads();
  const float e = __expf(s - mx);
  red[tid] = e; __syncthreads();
  for (int off = 128; off > 0; off >>= 1) {
    if (tid < off) red[tid] += red[tid + off];
    __syncthreads();
  }
  p[((size_t)b * 32 + h) * 256 + tid] = e * (1.0f / red[0]);
}

// c_n[b,h,d] = sum_t p[b,h,t] * wenc_n[b,t,d]   (coalesced in d)
__global__ void k_cn(const float* __restrict__ p, const float* __restrict__ q,
                     float* __restrict__ out) {
  size_t gid = (size_t)blockIdx.x * blockDim.x + threadIdx.x;
  if (gid >= (size_t)C_B * C_NH * C_HS) return;
  const int d = (int)(gid & 255);
  const int h = (int)((gid >> 8) & 31);
  const int b = (int)(gid >> 13);
  const float* pp = p + ((size_t)b * 32 + h) * 256;
  const float* qq = q + (size_t)b * 256 * 256 + d;
  float acc = 0.0f;
  for (int t = 0; t < 256; ++t) acc += pp[t] * qq[(size_t)t * 256];
  out[gid] = acc;
}

// score[b,h] = tanh(concat(y1,y2)) . W_out + b_out, with l_hs penalty mask
__global__ void k_score(const float* __restrict__ y1, const float* __restrict__ y2,
                        const float* __restrict__ Wout, const float* __restrict__ bout,
                        const int* __restrict__ l_hs, float* __restrict__ out) {
  const int gid = blockIdx.x * blockDim.x + threadIdx.x;
  if (gid >= C_B * C_NH) return;
  const int b = gid >> 5, h = gid & 31;
  const float* a1 = y1 + (size_t)gid * 256;
  const float* a2 = y2 + (size_t)gid * 256;
  float acc = bout[0];
  for (int d = 0; d < 256; ++d)
    acc += tanhf(a1[d]) * Wout[d] + tanhf(a2[d]) * Wout[256 + d];
  out[gid] = (h < l_hs[b]) ? acc : NEGV;
}

// ---------------------------------------------------------------------------
// Host orchestration
// ---------------------------------------------------------------------------
struct PackedEnc {
  __bf16* wih[2][2];
  __bf16* whh[2][2];
  float*  bias[2][2];
};

static inline const float* leaf(void* const* d_in, int base, int layer, int dir, int which) {
  return (const float*)d_in[base + layer * 8 + dir * 4 + which];
}

static inline dim3 g1(size_t n, int bs = 256) { return dim3((unsigned)((n + bs - 1) / bs)); }

extern "C" void kernel_launch(void* const* d_in, const int* in_sizes, int n_in,
                              void* d_out, int out_size, void* d_ws, size_t ws_size,
                              hipStream_t stream) {
  (void)in_sizes; (void)n_in; (void)out_size; (void)ws_size;
  char* ws = (char*)d_ws;

  const float* wemb_n   = (const float*)d_in[0];
  const int*   l_n      = (const int*)d_in[1];
  const float* wemb_hpu = (const float*)d_in[2];
  const int*   l_hpu    = (const int*)d_in[3];
  const int*   l_hs     = (const int*)d_in[4];
  // enc_n_params leaves: 5..20, enc_h_params leaves: 21..36
  const float* W_att = (const float*)d_in[37];
  const float* b_att = (const float*)d_in[38];
  const float* W_c   = (const float*)d_in[39];
  const float* b_c   = (const float*)d_in[40];
  const float* W_hs  = (const float*)d_in[41];
  const float* b_hs  = (const float*)d_in[42];
  const float* W_out = (const float*)d_in[43];
  const float* b_out = (const float*)d_in[44];

  // ---- pack weights (bump-allocated inside PACK region) ----
  __bf16* packBase = (__bf16*)(ws + OFF_PACK);
  size_t packOfs = 0;
  auto packW = [&](const float* W, int K, int Nout) -> __bf16* {
    __bf16* dst = packBase + packOfs;
    const size_t total = (size_t)K * Nout;
    k_pack_w<<<g1(total), 256, 0, stream>>>(W, dst, K, Nout);
    packOfs += total;
    return dst;
  };

  PackedEnc pe[2];
  float* bptr = (float*)(ws + OFF_BIAS);
  const int bases[2] = {5, 21};   // enc_n, enc_h
  for (int e = 0; e < 2; ++e)
    for (int l = 0; l < 2; ++l)
      for (int d = 0; d < 2; ++d) {
        const int D = l ? C_HS : C_IS;
        pe[e].wih[l][d] = packW(leaf(d_in, bases[e], l, d, 0), D, 4 * C_H);
        pe[e].whh[l][d] = packW(leaf(d_in, bases[e], l, d, 1), C_H, 4 * C_H);
        pe[e].bias[l][d] = bptr;
        k_bias2<<<2, 256, 0, stream>>>(leaf(d_in, bases[e], l, d, 2),
                                       leaf(d_in, bases[e], l, d, 3), bptr, 512);
        bptr += 512;
      }
  __bf16* Watt_p = packW(W_att, C_HS, C_HS);
  __bf16* Wc_p   = packW(W_c,   C_HS, C_HS);
  __bf16* Whs_p  = packW(W_hs,  C_HS, C_HS);

  // ---- shared scratch pointers ----
  __bf16* Xbf = (__bf16*)(ws + OFF_XBF);
  float* gF = (float*)(ws + OFF_GF);
  float* gB = (float*)(ws + OFF_GB);
  float* yA = (float*)(ws + OFF_YA);
  float* yB = (float*)(ws + OFF_YB);
  float* hF = (float*)(ws + OFF_STATE);
  float* cF = hF + (size_t)C_NHP * C_H;
  float* hB = cF + (size_t)C_NHP * C_H;
  float* cB = hB + (size_t)C_NHP * C_H;

  auto run_enc = [&](const float* Xin, const int* lens, const PackedEnc& P,
                     int N, int T) {
    for (int l = 0; l < 2; ++l) {
      const int D = l ? C_HS : C_IS;
      const size_t rows = (size_t)N * T;
      // 1) big bf16 WMMA GEMM: pre-gates = X @ Wih^T + (bih+bhh), both dirs
      if (l == 0) k_cvt_bf16<<<g1(rows * C_IS), 256, 0, stream>>>(Xin, Xbf, rows * C_IS);
      else        k_cvt_bf16<<<g1(rows * C_HS), 256, 0, stream>>>(yA, Xbf, rows * C_HS);
      dim3 gg((unsigned)(rows / 64), 2);   // 64-row tiles, 512 gate cols = 2x256
      k_gemm<<<gg, 128, 0, stream>>>(Xbf, P.wih[l][0], P.bias[l][0], gF,
                                     (int)rows, D, 512);
      k_gemm<<<gg, 128, 0, stream>>>(Xbf, P.wih[l][1], P.bias[l][1], gB,
                                     (int)rows, D, 512);
      // 2) recurrence
      float* y = l ? yB : yA;
      k_zero_f32<<<g1(rows * C_HS), 256, 0, stream>>>(y, rows * C_HS);
      k_zero_f32<<<g1((size_t)4 * C_NHP * C_H), 256, 0, stream>>>(hF, (size_t)4 * C_NHP * C_H);
      for (int t = 0; t < T; ++t)
        k_lstm_step<<<dim3((unsigned)(N / 16), 2), 256, 0, stream>>>(
            gF, gB, P.whh[l][0], P.whh[l][1], hF, cF, hB, cB, lens, y, N, T, t);
    }
  };

  // ---- header encoder first (its y buffers get reused by the question enc)
  run_enc(wemb_hpu, l_hpu, pe[1], C_NHP, C_TH);
  float* wenc_hs = (float*)(ws + OFF_HS);
  k_gather_hs<<<g1((size_t)C_NHP * C_HS), 256, 0, stream>>>(yB, l_hpu, wenc_hs);

  // ---- question encoder; final output stays in yB ([256,256,256])
  run_enc(wemb_n, l_n, pe[0], C_B, C_TN);
  float* wenc_n = yB;

  // ---- attention projection: (wenc_n @ W_att^T + b_att) via WMMA
  const size_t qrows = (size_t)C_B * C_TN;
  k_cvt_bf16<<<g1(qrows * C_HS), 256, 0, stream>>>(wenc_n, Xbf, qrows * C_HS);
  float* attproj = gF;  // reuse gate scratch
  k_gemm<<<dim3((unsigned)(qrows / 64), 1), 128, 0, stream>>>(
      Xbf, Watt_p, b_att, attproj, (int)qrows, C_HS, C_HS);

  // ---- masked softmax attention + context
  float* p   = (float*)(ws + OFF_P);
  float* c_n = (float*)(ws + OFF_CN);
  k_att_softmax<<<dim3(C_B * C_NH), 256, 0, stream>>>(wenc_hs, attproj, l_n, p);
  k_cn<<<g1((size_t)C_B * C_NH * C_HS), 256, 0, stream>>>(p, wenc_n, c_n);

  // ---- output linears (WMMA) + scoring
  float* y1 = (float*)(ws + OFF_Y1);
  float* y2 = (float*)(ws + OFF_Y2);
  const size_t hrows = (size_t)C_B * C_NH;  // 8192
  k_cvt_bf16<<<g1(hrows * C_HS), 256, 0, stream>>>(c_n, Xbf, hrows * C_HS);
  k_gemm<<<dim3((unsigned)(hrows / 64), 1), 128, 0, stream>>>(
      Xbf, Wc_p, b_c, y1, (int)hrows, C_HS, C_HS);
  k_cvt_bf16<<<g1(hrows * C_HS), 256, 0, stream>>>(wenc_hs, Xbf, hrows * C_HS);
  k_gemm<<<dim3((unsigned)(hrows / 64), 1), 128, 0, stream>>>(
      Xbf, Whs_p, b_hs, y2, (int)hrows, C_HS, C_HS);

  k_score<<<dim3(C_B * C_NH / 256), 256, 0, stream>>>(
      y1, y2, W_out, b_out, l_hs, (float*)d_out);
}
